// AttnEncoder_7267084665222
// MI455X (gfx1250) — compile-verified, matrix operand order checked by reading, not run on hardware
//
#include <hip/hip_runtime.h>
#include <hip/hip_bf16.h>

// Problem constants (match reference).
#define B_  256
#define T_  128
#define I_  256
#define H_  256
#define G4  1024  // 4*H

typedef __attribute__((ext_vector_type(16))) __bf16 v16bf;
typedef __attribute__((ext_vector_type(8)))  float  v8f;

union Frag {
  v16bf v;
  uint4 q[2];
};

// float -> bf16 round-to-nearest-even (bit pattern as ushort)
__device__ __forceinline__ unsigned short f2bf(float f) {
  union { float f; unsigned u; } c; c.f = f;
  unsigned u = c.u;
  unsigned r = (u + 0x7FFFu + ((u >> 16) & 1u)) >> 16;
  return (unsigned short)r;
}

// Load one lane's 16-element bf16 fragment for a 16x32 (or 32x16) tile.
// p16 points at (row * ld + kc*32) in a bf16(ushort) array, 16B-aligned.
// Lanes 0-15 (sel=0) hold K {0..7,16..23}; lanes 16-31 (sel=1) hold K {8..15,24..31}.
__device__ __forceinline__ v16bf load_frag(const unsigned short* p16, int sel) {
  Frag f;
  const uint4* p = reinterpret_cast<const uint4*>(p16);
  f.q[0] = p[sel];
  f.q[1] = p[sel + 2];
  return f.v;
}

#define WMMA_BF16(A, Bm, C) \
  __builtin_amdgcn_wmma_f32_16x16x32_bf16(false, (A), false, (Bm), (short)0, (C), false, false)

// ---------------------------------------------------------------------------
// Kernel 0: weight conversion fp32->bf16, h0->bf16, bias combine, counter reset
// ---------------------------------------------------------------------------
__global__ void __launch_bounds__(256)
prep_kernel(const float* __restrict__ w_ih, const float* __restrict__ w_hh,
            const float* __restrict__ h0,   const float* __restrict__ b_ih,
            const float* __restrict__ b_hh,
            unsigned short* __restrict__ wih_bf, unsigned short* __restrict__ whh_bf,
            unsigned short* __restrict__ h_buf0, float* __restrict__ bias,
            unsigned* __restrict__ ctr) {
  int idx = blockIdx.x * 256 + threadIdx.x;
  if (idx < G4 * I_) {
    wih_bf[idx] = f2bf(w_ih[idx]);
    whh_bf[idx] = f2bf(w_hh[idx]);
  }
  if (idx < B_ * H_) h_buf0[idx] = f2bf(h0[idx]);
  if (idx < G4)      bias[idx]   = b_ih[idx] + b_hh[idx];
  if (idx < 64)      ctr[idx]    = 0;   // reset persistent-kernel barrier counters
}

// ---------------------------------------------------------------------------
// Kernel 1: attention. a_t is t-invariant (softmax shift-invariance: the
// h*w_h + c*w_c term is a per-batch scalar, constant along the softmax axis,
// and b_attn is a global scalar -> both cancel). One softmax per batch row.
// Also precomputes wi[t][b][i] = a[b,i] * x[b,t,i] in bf16 for the GEMMs.
// ---------------------------------------------------------------------------
__global__ void __launch_bounds__(256)
attn_kernel(const float* __restrict__ x, const float* __restrict__ w_attn,
            float* __restrict__ attn_out, unsigned short* __restrict__ wi_bf) {
  int b = blockIdx.x;
  int i = threadIdx.x;                       // one thread per input channel
  const float* ws = w_attn + 2 * H_;         // w_s = w_attn[0, 2H:]
  const float* xb = x + (size_t)b * T_ * I_;

  float acc = 0.f;                           // e_series[b,i] = sum_t x[b,t,i]*w_s[t]
  for (int t = 0; t < T_; ++t) acc += xb[t * I_ + i] * ws[t];

  __shared__ float red[256];
  red[i] = acc; __syncthreads();
  for (int s = 128; s > 0; s >>= 1) {        // max reduce
    if (i < s) red[i] = fmaxf(red[i], red[i + s]);
    __syncthreads();
  }
  float m = red[0]; __syncthreads();
  float p = __expf(acc - m);
  red[i] = p; __syncthreads();
  for (int s = 128; s > 0; s >>= 1) {        // sum reduce
    if (i < s) red[i] += red[i + s];
    __syncthreads();
  }
  float a = p / red[0];

  float* ao = attn_out + (size_t)b * T_ * I_;
  for (int t = 0; t < T_; ++t) {
    float xv = xb[t * I_ + i];
    ao[t * I_ + i] = a;                                       // attns output (broadcast over t)
    wi_bf[((size_t)t * B_ + b) * I_ + i] = f2bf(a * xv);      // GEMM A operand, K-contiguous
  }
}

// ---------------------------------------------------------------------------
// Kernel 2: input-gate pre-activations for ALL steps (fully parallel, no
// recurrence): gin[t][b][n] = sum_i wi[t,b,i]*W_ih[n,i] + b_ih[n] + b_hh[n].
// Row space = (t,b) pairs: 32768 x 1024, K=256. Stored fp16 (range-safe).
// Block = 8 waves; wave w owns row-tile (blockIdx.x*8 + w), two col-tiles.
// ---------------------------------------------------------------------------
__global__ void __launch_bounds__(256)
gin_kernel(const unsigned short* __restrict__ wi_bf,
           const unsigned short* __restrict__ wih_bf,
           const float* __restrict__ bias_g,
           _Float16* __restrict__ gin) {
  const int tid  = threadIdx.x;
  const int wave = tid >> 5;
  const int lane = tid & 31;
  const int lrow = lane & 15;
  const int sel  = (lane < 16) ? 0 : 1;

  const int rt  = blockIdx.x * 8 + wave;   // row tile: 0..2047
  const int ct0 = blockIdx.y * 2;          // col tiles: ct0, ct0+1 (0..63)

  const unsigned short* ap  = wi_bf  + ((size_t)rt * 16 + lrow) * I_;
  const unsigned short* b0p = wih_bf + (size_t)(ct0 * 16 + lrow) * I_;
  const unsigned short* b1p = wih_bf + (size_t)((ct0 + 1) * 16 + lrow) * I_;

  v16bf af[8];
  #pragma unroll
  for (int kc = 0; kc < 8; ++kc) af[kc] = load_frag(ap + kc * 32, sel);

  v8f acc0 = {};
  v8f acc1 = {};
  #pragma unroll
  for (int kc = 0; kc < 8; ++kc) {
    v16bf w0 = load_frag(b0p + kc * 32, sel);
    v16bf w1 = load_frag(b1p + kc * 32, sel);
    acc0 = WMMA_BF16(af[kc], w0, acc0);
    acc1 = WMMA_BF16(af[kc], w1, acc1);
  }

  const float bias0 = bias_g[ct0 * 16 + lrow];
  const float bias1 = bias_g[(ct0 + 1) * 16 + lrow];
  const int mbase = sel * 8;
  const size_t rowbase = (size_t)rt * 16;
  #pragma unroll
  for (int r = 0; r < 8; ++r) {
    gin[(rowbase + mbase + r) * G4 + ct0 * 16 + lrow]       = (_Float16)(acc0[r] + bias0);
    gin[(rowbase + mbase + r) * G4 + (ct0 + 1) * 16 + lrow] = (_Float16)(acc1[r] + bias1);
  }
}

// ---------------------------------------------------------------------------
// Kernel 3: persistent LSTM recurrence. 64 blocks = 16 batch-chunks x 4
// hidden-column chunks. Per step each block computes its 16x256 gate slab as
// gin[t] (prefetched into registers a step ahead) + h*W_hh^T via bf16 WMMA,
// runs the fp32 cell, writes h to output + bf16 ping-pong buffer, then a
// 4-block global barrier makes h visible for the next step.
//
// W_hh fragments are VGPR-resident for all 128 steps; only ~128 VGPRs, which
// leaves room for all 8 h-fragments to stay live -> ONE batched load + ONE
// wait on the recurrent critical path per step.
// ---------------------------------------------------------------------------
__global__ void __launch_bounds__(256)
lstm_kernel(const _Float16* __restrict__ gin,
            const unsigned short* __restrict__ whh_bf,
            const float* __restrict__ c0,
            unsigned short* __restrict__ hbuf,   // [2][B_*H_] bf16 ping-pong
            float* __restrict__ enc_out,
            unsigned* __restrict__ ctr) {
  const int bb   = blockIdx.x & 15;     // batch chunk: rows [16*bb, 16*bb+16)
  const int nn   = blockIdx.x >> 4;     // hidden-col chunk: cols [64*nn, 64*nn+64)
  const int b0   = bb * 16;
  const int tid  = threadIdx.x;
  const int wave = tid >> 5;
  const int lane = tid & 31;
  const int lrow = lane & 15;
  const int sel  = (lane < 16) ? 0 : 1;

  __shared__ float gates[16][272];      // 16 batch rows x 256 gate cols (+pad)
  __shared__ float c_s[16][64];         // persistent cell state for this block

  for (int e = tid; e < 16 * 64; e += 256) {
    int m = e >> 6, hl = e & 63;
    c_s[m][hl] = c0[(size_t)(b0 + m) * H_ + nn * 64 + hl];
  }
  __syncthreads();

  // Each wave owns two 16x16 output tiles tt0, tt0+1 of the 16 tiles.
  // tile tt -> gate g = tt>>2, sub = tt&3 -> weight row n = g*256 + nn*64 + sub*16 + lrow
  const int tt0 = wave * 2;
  const int g0 = tt0 >> 2,        s0 = tt0 & 3;
  const int g1 = (tt0 + 1) >> 2,  s1 = (tt0 + 1) & 3;
  const int n0 = g0 * H_ + nn * 64 + s0 * 16 + lrow;
  const int n1 = g1 * H_ + nn * 64 + s1 * 16 + lrow;
  const unsigned short* bh0 = whh_bf + (size_t)n0 * H_;
  const unsigned short* bh1 = whh_bf + (size_t)n1 * H_;

  // W_hh fragments: resident in VGPRs for the whole recurrence (2 x 8 x 8 VGPRs).
  v16bf u0f[8], u1f[8];
  #pragma unroll
  for (int kc = 0; kc < 8; ++kc) {
    u0f[kc] = load_frag(bh0 + kc * 32, sel);
    u1f[kc] = load_frag(bh1 + kc * 32, sel);
  }

  // Prefetch gin for step 0 into registers (raw fp16; converted in the cell).
  _Float16 gv[16];
  #pragma unroll
  for (int e2 = 0; e2 < 4; ++e2) {
    int e = tid + e2 * 256;
    int m = e >> 6, hl = e & 63;
    const _Float16* gp = gin + ((size_t)(b0 + m)) * G4 + nn * 64 + hl;
    gv[e2 * 4 + 0] = gp[0];
    gv[e2 * 4 + 1] = gp[256];
    gv[e2 * 4 + 2] = gp[512];
    gv[e2 * 4 + 3] = gp[768];
  }

  unsigned* myctr = ctr + bb;

  for (int t = 0; t < T_; ++t) {
    const unsigned short* hrd = hbuf + (size_t)(t & 1) * (B_ * H_);
    unsigned short*       hwr = hbuf + (size_t)((t + 1) & 1) * (B_ * H_);

    // h fragments for this step: issue all 8 loads as one clause -> one wait.
    const unsigned short* a2b = hrd + (size_t)(b0 + lrow) * H_;
    v16bf a2f[8];
    #pragma unroll
    for (int kc = 0; kc < 8; ++kc) a2f[kc] = load_frag(a2b + kc * 32, sel);

    v8f acc0 = {};
    v8f acc1 = {};
    #pragma unroll
    for (int kc = 0; kc < 8; ++kc) {
      acc0 = WMMA_BF16(a2f[kc], u0f[kc], acc0);
      acc1 = WMMA_BF16(a2f[kc], u1f[kc], acc1);
    }

    // C/D layout: lanes 0-15 -> M=r, N=lane; lanes 16-31 -> M=8+r, N=lane-16.
    const int mbase = sel * 8;
    #pragma unroll
    for (int r = 0; r < 8; ++r) {
      gates[mbase + r][tt0 * 16 + lrow]       = acc0[r];
      gates[mbase + r][(tt0 + 1) * 16 + lrow] = acc1[r];
    }
    __syncthreads();

    // LSTM cell (fp32). LDS gate col for (g, hl) is g*64 + hl.
    #pragma unroll
    for (int e2 = 0; e2 < 4; ++e2) {
      int e = tid + e2 * 256;
      int m = e >> 6, hl = e & 63;
      float ig = gates[m][0 * 64 + hl] + (float)gv[e2 * 4 + 0];
      float fg = gates[m][1 * 64 + hl] + (float)gv[e2 * 4 + 1];
      float gg = gates[m][2 * 64 + hl] + (float)gv[e2 * 4 + 2];
      float og = gates[m][3 * 64 + hl] + (float)gv[e2 * 4 + 3];
      float si = 1.f / (1.f + __expf(-ig));
      float sf = 1.f / (1.f + __expf(-fg));
      float so = 1.f / (1.f + __expf(-og));
      float cn = sf * c_s[m][hl] + si * tanhf(gg);
      c_s[m][hl] = cn;
      float hn = so * tanhf(cn);
      int gb = b0 + m;
      enc_out[((size_t)gb * T_ + t) * H_ + nn * 64 + hl] = hn;
      hwr[(size_t)gb * H_ + nn * 64 + hl] = f2bf(hn);
    }

    // Early-issue next step's gin prefetch: latency hides behind fence+barrier.
    if (t + 1 < T_) {
      #pragma unroll
      for (int e2 = 0; e2 < 4; ++e2) {
        int e = tid + e2 * 256;
        int m = e >> 6, hl = e & 63;
        const _Float16* gp = gin + ((size_t)(t + 1) * B_ + b0 + m) * G4 + nn * 64 + hl;
        gv[e2 * 4 + 0] = gp[0];
        gv[e2 * 4 + 1] = gp[256];
        gv[e2 * 4 + 2] = gp[512];
        gv[e2 * 4 + 3] = gp[768];
      }
    }

    __syncthreads();
    __threadfence();                 // make h stores visible at device scope

    // 4-block arrive + spin barrier per batch chunk (blocks bb, bb+16, bb+32, bb+48)
    if (tid == 0) {
      __hip_atomic_fetch_add(myctr, 1u, __ATOMIC_ACQ_REL, __HIP_MEMORY_SCOPE_AGENT);
      const unsigned target = 4u * (unsigned)(t + 1);
      while (__hip_atomic_load(myctr, __ATOMIC_ACQUIRE, __HIP_MEMORY_SCOPE_AGENT) < target)
        __builtin_amdgcn_s_sleep(2);
    }
    __syncthreads();
    __threadfence();
  }
}

// ---------------------------------------------------------------------------
extern "C" void kernel_launch(void* const* d_in, const int* in_sizes, int n_in,
                              void* d_out, int out_size, void* d_ws, size_t ws_size,
                              hipStream_t stream) {
  (void)in_sizes; (void)n_in; (void)out_size; (void)ws_size;
  const float* x      = (const float*)d_in[0];  // (B,T,I)
  const float* h0     = (const float*)d_in[1];  // (1,B,H)
  const float* c0     = (const float*)d_in[2];  // (1,B,H)
  const float* w_attn = (const float*)d_in[3];  // (1,2H+T)
  // d_in[4] = b_attn: cancels through softmax shift-invariance.
  const float* w_ih   = (const float*)d_in[5];  // (4H,I)
  const float* w_hh   = (const float*)d_in[6];  // (4H,H)
  const float* b_ih   = (const float*)d_in[7];
  const float* b_hh   = (const float*)d_in[8];

  float* attns = (float*)d_out;                         // (B,T,I)
  float* enc   = (float*)d_out + (size_t)B_ * T_ * I_;  // (B,T,H)

  // Workspace layout (bytes), ~85.2 MB total.
  char* ws = (char*)d_ws;
  unsigned short* wih_bf = (unsigned short*)(ws + 0);          // 512 KB
  unsigned short* whh_bf = (unsigned short*)(ws + 524288);     // 512 KB
  float*          bias   = (float*)         (ws + 1048576);    // 4 KB
  unsigned short* hbuf   = (unsigned short*)(ws + 1052672);    // 256 KB (2 x B x H bf16)
  unsigned short* wi_bf  = (unsigned short*)(ws + 1314816);    // 16 MB (T x B x I bf16)
  _Float16*       gin    = (_Float16*)      (ws + 18092032);   // 64 MB (T x B x 4H fp16)
  unsigned*       ctr    = (unsigned*)      (ws + 85200896);   // 256 B

  prep_kernel<<<(G4 * I_) / 256, 256, 0, stream>>>(
      w_ih, w_hh, h0, b_ih, b_hh, wih_bf, whh_bf, hbuf, bias, ctr);

  attn_kernel<<<B_, 256, 0, stream>>>(x, w_attn, attns, wi_bf);

  dim3 ggrid(256, 32);
  gin_kernel<<<ggrid, 256, 0, stream>>>(wi_bf, wih_bf, bias, gin);

  lstm_kernel<<<64, 256, 0, stream>>>(gin, whh_bf, c0, hbuf, enc, ctr);
}
